// TransformerLayer_21955872817165
// MI455X (gfx1250) — matrix operand order repaired and checked by reading, not compiled, not run on hardware
//
#include <hip/hip_runtime.h>
#include <hip/hip_bf16.h>

typedef __attribute__((ext_vector_type(16))) _Float16 v16h;
typedef __attribute__((ext_vector_type(8)))  float    v8f;

#define D_MODEL 1024
#define N_HEADS 16
#define HEAD_DIM 64
#define D_FF    4096
#define BATCH   2
#define SEQ     2048
#define BS      (BATCH*SEQ)

// ---------------------------------------------------------------------------
// CDNA5 async global->LDS (ASYNCcnt path), with portable fallback
// Builtin signature (probe-confirmed via diagnostic): (int4 AS1*, int4 AS3*, Ii, Ii)
// ---------------------------------------------------------------------------
#if defined(__has_builtin)
# if __has_builtin(__builtin_amdgcn_global_load_async_to_lds_b128)
#  define HAVE_ASYNC_LDS 1
# endif
#endif
#ifndef HAVE_ASYNC_LDS
# define HAVE_ASYNC_LDS 0
#endif

typedef int vint4 __attribute__((vector_size(16)));
typedef __attribute__((address_space(1))) vint4 as1_vint4;
typedef __attribute__((address_space(3))) vint4 as3_vint4;

__device__ __forceinline__ void copy16B_to_lds(const _Float16* g, _Float16* l) {
#if HAVE_ASYNC_LDS
  __builtin_amdgcn_global_load_async_to_lds_b128(
      (as1_vint4*)g, (as3_vint4*)l, 0, 0);
#else
  *(uint4*)l = *(const uint4*)g;
#endif
}
__device__ __forceinline__ void wait_async0() {
#if HAVE_ASYNC_LDS
# if __has_builtin(__builtin_amdgcn_s_wait_asynccnt)
  __builtin_amdgcn_s_wait_asynccnt(0);
# else
  asm volatile("s_wait_asynccnt 0x0" ::: "memory");
# endif
#endif
}

// ---------------------------------------------------------------------------
// WMMA fragment helpers (CDNA5 16x16x32 f16, wave32)
// A (16x32, [M][K] source, K contiguous): lane holds row m=lane&15,
//   elems 0..15 -> k = half*16+i                               (two 16B loads)
// B (32x16, sourced from B^T [N][K], K contiguous): lane holds col n=lane&15,
//   elems 0..15 -> k = half*16+i                               (two 16B loads)
// ---------------------------------------------------------------------------
__device__ __forceinline__ v16h ld_a_frag(const _Float16* row, int hlf) {
  union { v16h v; uint4 u[2]; } r;
  const char* p = (const char*)row + hlf * 32;   // half*16 halves
  r.u[0] = *(const uint4*)(p);
  r.u[1] = *(const uint4*)(p + 16);              // +8 halves
  return r.v;
}
__device__ __forceinline__ v16h ld_b_frag(const _Float16* row, int hlf) {
  union { v16h v; uint4 u[2]; } r;
  const uint4* p = (const uint4*)((const char*)row + hlf * 32); // half*16 halves
  r.u[0] = p[0];
  r.u[1] = p[1];
  return r.v;
}
__device__ __forceinline__ v8f wmma_f16(v16h a, v16h b, v8f c) {
  return __builtin_amdgcn_wmma_f32_16x16x32_f16(
      /*neg_a=*/false, a, /*neg_b=*/false, b,
      /*c_mod=*/(short)0, c, /*reuse_a=*/false, /*reuse_b=*/false);
}
__device__ __forceinline__ v8f vzero8() {
  v8f z = {0.f,0.f,0.f,0.f,0.f,0.f,0.f,0.f};
  return z;
}

// ---------------------------------------------------------------------------
// Cast / packing kernels (run once per call; all weights -> f16 [N][K])
// ---------------------------------------------------------------------------
__global__ __launch_bounds__(256)
void cast_f16_kernel(const float* __restrict__ in, _Float16* __restrict__ out, int n) {
  int i = blockIdx.x * 256 + threadIdx.x;
  if (i < n) out[i] = (_Float16)in[i];
}

// Wq/Wk/Wv: [H][D][hd] -> packed B^T layout [c=h*64+k][d]  (N x K, K=D contig)
__global__ __launch_bounds__(256)
void pack_headw_kernel(const float* __restrict__ in, _Float16* __restrict__ out) {
  int i = blockIdx.x * 256 + threadIdx.x;          // over D*D
  if (i >= D_MODEL * D_MODEL) return;
  int c = i >> 10, d = i & 1023;
  int h = c >> 6,  k = c & 63;
  out[i] = (_Float16)in[((size_t)h * D_MODEL + d) * HEAD_DIM + k];
}

// [K][N] f32 -> [N][K] f16
__global__ __launch_bounds__(256)
void transpose_cast_kernel(const float* __restrict__ in, _Float16* __restrict__ out,
                           int Kd, int Nd) {
  int i = blockIdx.x * 256 + threadIdx.x;          // over Kd*Nd (out index)
  if (i >= Kd * Nd) return;
  int n = i / Kd, k = i - n * Kd;
  out[i] = (_Float16)in[(size_t)k * Nd + n];
}

// ---------------------------------------------------------------------------
// Tiled WMMA GEMM: C[M][N] = A[M][K] * Bt[N][K]^T + bias (+resid / relu)
// Block 256 threads (8 waves), tile 128x128x32. Waves 2(M) x 4(N), each 64x32.
// Double-buffered LDS; async global->LDS staging overlaps WMMA compute.
// EPI: 0 = bias -> f16 ; 1 = bias + resid -> f32 ; 2 = bias + relu -> f16
// ---------------------------------------------------------------------------
template<int EPI>
__global__ __launch_bounds__(256, 2)
void gemm_kernel(const _Float16* __restrict__ A, const _Float16* __restrict__ Bt,
                 const float* __restrict__ bias, const float* __restrict__ resid,
                 void* __restrict__ out, int M, int N, int K) {
  constexpr int BM = 128, BN = 128, BK = 32, LS = 40; // LDS row stride (pad, 16B-aligned)
  __shared__ _Float16 sA[2][BM * LS];
  __shared__ _Float16 sB[2][BN * LS];

  const int tid  = threadIdx.x;
  const int lane = tid & 31, wave = tid >> 5;
  const int hlf  = lane >> 4, l15 = lane & 15;
  const int wm   = wave & 1,  wn  = wave >> 1;
  const int m0   = blockIdx.y * BM, n0 = blockIdx.x * BN;

  v8f acc[4][2];
#pragma unroll
  for (int mt = 0; mt < 4; ++mt)
#pragma unroll
    for (int nt = 0; nt < 2; ++nt) acc[mt][nt] = vzero8();

  const int cra = tid >> 1, csa = tid & 1;   // copy: row, 16-half segment
  const int nk = K / BK;

  auto stage = [&](int kt, int bi) {
    const _Float16* ga = A  + (size_t)(m0 + cra) * K + kt * BK + csa * 16;
    const _Float16* gb = Bt + (size_t)(n0 + cra) * K + kt * BK + csa * 16;
    _Float16* la = &sA[bi][cra * LS + csa * 16];
    _Float16* lb = &sB[bi][cra * LS + csa * 16];
    copy16B_to_lds(ga,     la);
    copy16B_to_lds(ga + 8, la + 8);
    copy16B_to_lds(gb,     lb);
    copy16B_to_lds(gb + 8, lb + 8);
  };

  stage(0, 0);
  wait_async0();
  __syncthreads();

  for (int kt = 0; kt < nk; ++kt) {
    const int bi = kt & 1;
    if (kt + 1 < nk) stage(kt + 1, bi ^ 1);   // prefetch next tile (ASYNCcnt path)

    v16h af[4], bf[2];
#pragma unroll
    for (int mt = 0; mt < 4; ++mt)
      af[mt] = ld_a_frag(&sA[bi][(wm * 64 + mt * 16 + l15) * LS], hlf);
#pragma unroll
    for (int nt = 0; nt < 2; ++nt)
      bf[nt] = ld_b_frag(&sB[bi][(wn * 32 + nt * 16 + l15) * LS], hlf);
#pragma unroll
    for (int mt = 0; mt < 4; ++mt)
#pragma unroll
      for (int nt = 0; nt < 2; ++nt)
        acc[mt][nt] = wmma_f16(af[mt], bf[nt], acc[mt][nt]);

    wait_async0();
    __syncthreads();
  }

  // epilogue: C layout -> VGPR r holds row (r + 8*half), col = tile base + l15
#pragma unroll
  for (int mt = 0; mt < 4; ++mt) {
#pragma unroll
    for (int nt = 0; nt < 2; ++nt) {
      const int col = n0 + wn * 32 + nt * 16 + l15;
      const float bv = bias[col];
#pragma unroll
      for (int r = 0; r < 8; ++r) {
        const int row = m0 + wm * 64 + mt * 16 + r + hlf * 8;
        float v = acc[mt][nt][r] + bv;
        if (EPI == 1) {
          v += resid[(size_t)row * N + col];
          ((float*)out)[(size_t)row * N + col] = v;
        } else {
          if (EPI == 2) v = v > 0.f ? v : 0.f;
          ((_Float16*)out)[(size_t)row * N + col] = (_Float16)v;
        }
      }
    }
  }
}

// ---------------------------------------------------------------------------
// Flash attention: block = (b, h, 64 q rows), 4 waves, 32-key tiles,
// online softmax, WMMA for Q*K^T and P*V. head_dim = 64.
// ---------------------------------------------------------------------------
__global__ __launch_bounds__(128, 2)
void attn_kernel(const _Float16* __restrict__ Q, const _Float16* __restrict__ Kmat,
                 const _Float16* __restrict__ V, _Float16* __restrict__ O) {
  constexpr int KT = 32;
  __shared__ _Float16 sK[32 * 72];        // [key][d], d contiguous (B^T for scores)
  __shared__ _Float16 sV[64 * 40];        // [d][key], key contiguous (B^T for P*V)
  __shared__ _Float16 sP[4 * 16 * 40];    // per-wave P staging (C-layout -> A-layout)

  const int tid  = threadIdx.x;
  const int lane = tid & 31, wave = tid >> 5;
  const int hlf  = lane >> 4, l15 = lane & 15;
  const int b = blockIdx.z, h = blockIdx.y;
  const int q0 = blockIdx.x * 64 + wave * 16;
  const size_t hoff = (size_t)h * HEAD_DIM;

  // Q fragments for this wave's 16 rows (d contiguous in global memory)
  const _Float16* qrow = Q + (size_t)(b * SEQ + q0 + l15) * D_MODEL + hoff;
  const v16h qa0 = ld_a_frag(qrow,      hlf);   // d 0..31
  const v16h qa1 = ld_a_frag(qrow + 32, hlf);   // d 32..63

  float mrow[8], lrow[8];
  v8f o[4];
#pragma unroll
  for (int r = 0; r < 8; ++r) { mrow[r] = -1e30f; lrow[r] = 0.f; }
#pragma unroll
  for (int t = 0; t < 4; ++t) o[t] = vzero8();

  _Float16* pw = sP + wave * 16 * 40;
  const float sc = 0.125f;                 // 1/sqrt(64)

  for (int j = 0; j < SEQ / KT; ++j) {
    const int kb = j * KT;
    { // K tile [32][64] -> sK (async global->LDS, stride 72)
      const int r = tid >> 2, seg = tid & 3;
      const _Float16* src = Kmat + (size_t)(b * SEQ + kb + r) * D_MODEL + hoff + seg * 16;
      _Float16* dst = sK + r * 72 + seg * 16;
      copy16B_to_lds(src,     dst);
      copy16B_to_lds(src + 8, dst + 8);
    }
    { // V tile transposed -> sV[d][key], stride 40 (manual transpose copy)
      const int r = tid & 31, dseg = tid >> 5;
      const _Float16* src = V + (size_t)(b * SEQ + kb + r) * D_MODEL + hoff + dseg * 16;
#pragma unroll
      for (int jj = 0; jj < 16; ++jj) sV[(dseg * 16 + jj) * 40 + r] = src[jj];
    }
    wait_async0();
    __syncthreads();

    // scores: 16 q rows x 32 keys = two 16x16 C tiles, chained over d (2x32)
    v8f s0 = vzero8(), s1 = vzero8();
    {
      const _Float16* k0 = sK + (l15)      * 72;
      const _Float16* k1 = sK + (16 + l15) * 72;
      s0 = wmma_f16(qa0, ld_b_frag(k0,      hlf), s0);
      s0 = wmma_f16(qa1, ld_b_frag(k0 + 32, hlf), s0);
      s1 = wmma_f16(qa0, ld_b_frag(k1,      hlf), s1);
      s1 = wmma_f16(qa1, ld_b_frag(k1 + 32, hlf), s1);
    }

    // online softmax: row r+8*half lives in VGPR r across 16 lanes
    float p0[8], p1[8];
#pragma unroll
    for (int r = 0; r < 8; ++r) {
      float v0 = s0[r] * sc, v1 = s1[r] * sc;
      float mx = fmaxf(v0, v1);
#pragma unroll
      for (int off = 1; off < 16; off <<= 1) mx = fmaxf(mx, __shfl_xor(mx, off, 32));
      const float mn = fmaxf(mrow[r], mx);
      const float al = __expf(mrow[r] - mn);
      const float e0 = __expf(v0 - mn), e1 = __expf(v1 - mn);
      float sm = e0 + e1;
#pragma unroll
      for (int off = 1; off < 16; off <<= 1) sm += __shfl_xor(sm, off, 32);
      lrow[r] = lrow[r] * al + sm;
      mrow[r] = mn;
      p0[r] = e0; p1[r] = e1;
#pragma unroll
      for (int t = 0; t < 4; ++t) o[t][r] *= al;
    }

    // stage P (C-layout) to LDS, reload as A fragment (K-dim = 32 keys)
#pragma unroll
    for (int r = 0; r < 8; ++r) {
      const int m = r + hlf * 8;
      pw[m * 40 + l15]      = (_Float16)p0[r];
      pw[m * 40 + 16 + l15] = (_Float16)p1[r];
    }
    const v16h pa = ld_a_frag(pw + l15 * 40, hlf);
#pragma unroll
    for (int t = 0; t < 4; ++t)
      o[t] = wmma_f16(pa, ld_b_frag(sV + (t * 16 + l15) * 40, hlf), o[t]);
    __syncthreads();
  }

  // normalize and write [B,S,D] f16
#pragma unroll
  for (int t = 0; t < 4; ++t)
#pragma unroll
    for (int r = 0; r < 8; ++r) {
      const int row = q0 + r + hlf * 8;
      O[(size_t)(b * SEQ + row) * D_MODEL + hoff + t * 16 + l15] =
          (_Float16)(o[t][r] / lrow[r]);
    }
}

// ---------------------------------------------------------------------------
// LayerNorm (block per row of 1024), optional f16 mirror for next GEMM
// ---------------------------------------------------------------------------
__global__ __launch_bounds__(256)
void layernorm_kernel(const float* __restrict__ in, const float* __restrict__ g,
                      const float* __restrict__ be, float* __restrict__ out32,
                      _Float16* __restrict__ out16) {
  __shared__ float red[8];
  __shared__ float sstat[2];
  const int row = blockIdx.x, tid = threadIdx.x;
  const int lane = tid & 31, wave = tid >> 5;
  const float* x = in + (size_t)row * D_MODEL;
  float v[4]; float s = 0.f;
#pragma unroll
  for (int k = 0; k < 4; ++k) { v[k] = x[tid + k * 256]; s += v[k]; }
#pragma unroll
  for (int off = 16; off > 0; off >>= 1) s += __shfl_xor(s, off, 32);
  if (lane == 0) red[wave] = s;
  __syncthreads();
  if (tid == 0) { float t = 0.f; for (int i = 0; i < 8; ++i) t += red[i];
                  sstat[0] = t * (1.f / D_MODEL); }
  __syncthreads();
  const float mu = sstat[0];
  float s2 = 0.f;
#pragma unroll
  for (int k = 0; k < 4; ++k) { const float d = v[k] - mu; s2 += d * d; }
#pragma unroll
  for (int off = 16; off > 0; off >>= 1) s2 += __shfl_xor(s2, off, 32);
  if (lane == 0) red[wave] = s2;
  __syncthreads();
  if (tid == 0) { float t = 0.f; for (int i = 0; i < 8; ++i) t += red[i];
                  sstat[1] = rsqrtf(t * (1.f / D_MODEL) + 1e-5f); }
  __syncthreads();
  const float rs = sstat[1];
#pragma unroll
  for (int k = 0; k < 4; ++k) {
    const int c = tid + k * 256;
    const float y = (v[k] - mu) * rs * g[c] + be[c];
    out32[(size_t)row * D_MODEL + c] = y;
    if (out16) out16[(size_t)row * D_MODEL + c] = (_Float16)y;
  }
}

// ---------------------------------------------------------------------------
extern "C" void kernel_launch(void* const* d_in, const int* in_sizes, int n_in,
                              void* d_out, int out_size, void* d_ws, size_t ws_size,
                              hipStream_t stream) {
  (void)in_sizes; (void)n_in; (void)out_size; (void)ws_size;
  const float* x   = (const float*)d_in[0];
  const float* Wq  = (const float*)d_in[1];
  const float* bq  = (const float*)d_in[2];
  const float* Wk  = (const float*)d_in[3];
  const float* bk  = (const float*)d_in[4];
  const float* Wv  = (const float*)d_in[5];
  const float* bv  = (const float*)d_in[6];
  const float* Wo  = (const float*)d_in[7];
  const float* bo  = (const float*)d_in[8];
  const float* W1  = (const float*)d_in[9];
  const float* b1  = (const float*)d_in[10];
  const float* W2  = (const float*)d_in[11];
  const float* b2  = (const float*)d_in[12];
  const float* g1  = (const float*)d_in[13];
  const float* be1 = (const float*)d_in[14];
  const float* g2  = (const float*)d_in[15];
  const float* be2 = (const float*)d_in[16];
  float* out = (float*)d_out;

  char* ws = (char*)d_ws;
  size_t off = 0;
  auto take = [&](size_t bytes) -> char* {
    char* p = ws + off; off += (bytes + 255) & ~(size_t)255; return p;
  };
  _Float16* xh  = (_Float16*)take((size_t)BS * D_MODEL * 2);   // x f16 (reused as h16)
  _Float16* wq  = (_Float16*)take((size_t)D_MODEL * D_MODEL * 2);
  _Float16* wk  = (_Float16*)take((size_t)D_MODEL * D_MODEL * 2);
  _Float16* wv  = (_Float16*)take((size_t)D_MODEL * D_MODEL * 2);
  _Float16* wo  = (_Float16*)take((size_t)D_MODEL * D_MODEL * 2);
  _Float16* w1  = (_Float16*)take((size_t)D_MODEL * D_FF * 2);
  _Float16* w2  = (_Float16*)take((size_t)D_MODEL * D_FF * 2);
  _Float16* qh  = (_Float16*)take((size_t)BS * D_MODEL * 2);   // reused: ff1 spans qh..ao
  _Float16* kh  = (_Float16*)take((size_t)BS * D_MODEL * 2);
  _Float16* vh  = (_Float16*)take((size_t)BS * D_MODEL * 2);
  _Float16* ao  = (_Float16*)take((size_t)BS * D_MODEL * 2);
  float*    ares= (float*)   take((size_t)BS * D_MODEL * 4);   // reused as ff2-res
  float*    h32 = (float*)   take((size_t)BS * D_MODEL * 4);
  _Float16* h16 = xh;                                          // alias (x consumed)
  _Float16* ff1 = qh;                                          // alias (qkv/ao consumed)
  float*    f2r = ares;                                        // alias (ares consumed)

  // 1) casts / weight packing
  cast_f16_kernel<<<(BS * D_MODEL) / 256, 256, 0, stream>>>(x, xh, BS * D_MODEL);
  pack_headw_kernel<<<(D_MODEL * D_MODEL) / 256, 256, 0, stream>>>(Wq, wq);
  pack_headw_kernel<<<(D_MODEL * D_MODEL) / 256, 256, 0, stream>>>(Wk, wk);
  pack_headw_kernel<<<(D_MODEL * D_MODEL) / 256, 256, 0, stream>>>(Wv, wv);
  transpose_cast_kernel<<<(D_MODEL * D_MODEL) / 256, 256, 0, stream>>>(Wo, wo, D_MODEL, D_MODEL);
  transpose_cast_kernel<<<(D_MODEL * D_FF) / 256, 256, 0, stream>>>(W1, w1, D_MODEL, D_FF);
  transpose_cast_kernel<<<(D_MODEL * D_FF) / 256, 256, 0, stream>>>(W2, w2, D_FF, D_MODEL);

  // 2) QKV projections (f16 out)
  dim3 gqkv(D_MODEL / 128, BS / 128);
  gemm_kernel<0><<<gqkv, 256, 0, stream>>>(xh, wq, bq, nullptr, qh, BS, D_MODEL, D_MODEL);
  gemm_kernel<0><<<gqkv, 256, 0, stream>>>(xh, wk, bk, nullptr, kh, BS, D_MODEL, D_MODEL);
  gemm_kernel<0><<<gqkv, 256, 0, stream>>>(xh, wv, bv, nullptr, vh, BS, D_MODEL, D_MODEL);

  // 3) flash attention -> ao (f16 [B,S,D])
  attn_kernel<<<dim3(SEQ / 64, N_HEADS, BATCH), 128, 0, stream>>>(qh, kh, vh, ao);

  // 4) output projection + residual(x) -> ares (f32)
  gemm_kernel<1><<<gqkv, 256, 0, stream>>>(ao, wo, bo, x, ares, BS, D_MODEL, D_MODEL);

  // 5) layernorm1 -> h32 / h16
  layernorm_kernel<<<BS, 256, 0, stream>>>(ares, g1, be1, h32, h16);

  // 6) FFN up + ReLU -> ff1 (f16 [BS, D_FF])
  gemm_kernel<2><<<dim3(D_FF / 128, BS / 128), 256, 0, stream>>>(
      h16, w1, b1, nullptr, ff1, BS, D_FF, D_MODEL);

  // 7) FFN down + residual(h32) -> f2r (f32)
  gemm_kernel<1><<<gqkv, 256, 0, stream>>>(ff1, w2, b2, h32, f2r, BS, D_MODEL, D_FF);

  // 8) layernorm2 -> out (f32)
  layernorm_kernel<<<BS, 256, 0, stream>>>(f2r, g2, be2, out, nullptr);
}
